// Swap_NET_43516608643102
// MI455X (gfx1250) — matrix operand order repaired and checked by reading, not compiled, not run on hardware
//
#include <hip/hip_runtime.h>

typedef __attribute__((ext_vector_type(16))) _Float16 v16h;
typedef __attribute__((ext_vector_type(8)))  _Float16 v8h;
typedef __attribute__((ext_vector_type(8)))  float    v8f;

#define B_   4
#define H_   256
#define W_   256
#define HW_  (H_*W_)
#define C_   64
#define K_   576          // 9 offsets * 64 channels
#define NG_  128          // stat partial groups per image
#define EPSF 1e-5f

__device__ __forceinline__ int refl(int i, int n) {
    return (i < 0) ? 1 : ((i >= n) ? (n - 2) : i);   // reflect pad width 1
}

// ---------------------------------------------------------------------------
// Pack the seven 3x3 weight tensors (OIHW f32) into f16 [64][576] matrices,
// koff = (ky*3+kx)*64 + c  so each K-chunk of 32 stays inside one offset.
// ---------------------------------------------------------------------------
__global__ void pack_weights_kernel(const float* w1a, const float* w1b,
                                    const float* w2a, const float* w2b,
                                    const float* w3a, const float* w3b,
                                    const float* wf,  _Float16* wpk)
{
    int idx = blockIdx.x * blockDim.x + threadIdx.x;
    const int per = C_ * K_;                     // 36864
    if (idx >= 7 * per) return;
    int s = idx / per, r = idx % per;
    int o = r / K_, koff = r % K_;
    int off = koff >> 6, c = koff & 63;
    const float* ws[7] = {w1a, w1b, w2a, w2b, w3a, w3b, wf};
    wpk[idx] = (_Float16)ws[s][(o * 64 + c) * 9 + off];
}

// h0 = w0 * x + b0  ->  f16 NHWC
__global__ void h0_kernel(const float* __restrict__ x, const float* __restrict__ w0,
                          const float* __restrict__ b0, _Float16* __restrict__ actA)
{
    int idx = blockIdx.x * blockDim.x + threadIdx.x;   // (b,pix,o) o fastest
    if (idx >= B_ * HW_ * C_) return;
    int o = idx & 63, bp = idx >> 6;
    actA[idx] = (_Float16)(w0[o] * x[bp] + b0[o]);
}

// kmap[b][off][pix] = exp(-0.5*(g(shift)-g(center))^2), f16
__global__ void kmap_kernel(const float* __restrict__ x, _Float16* __restrict__ kmap)
{
    int idx = blockIdx.x * blockDim.x + threadIdx.x;
    if (idx >= B_ * HW_) return;
    int b = idx / HW_, pix = idx % HW_;
    int y = pix / W_, xq = pix % W_;
    float gc = x[idx];
#pragma unroll
    for (int off = 0; off < 9; ++off) {
        int dy = off / 3 - 1, dx = off % 3 - 1;
        int yy = refl(y + dy, H_), xx = refl(xq + dx, W_);
        float d = x[b * HW_ + yy * W_ + xx] - gc;
        kmap[(b * 9 + off) * HW_ + pix] = (_Float16)__expf(-0.5f * d * d);
    }
}

// ---------------------------------------------------------------------------
// PAC conv via implicit GEMM on v_wmma_f32_16x16x32_f16.
// Block = 128 pixels (row segment) x all 64 out channels = 32 C-tiles.
// 8 waves, each owns a 2x2 tile grid: M-tiles {(w>>2)*2,+1} x N-tiles
// {(w&3)*2,+1}. Per K-chunk: 2 A-frags + 2 B-frags -> 4 WMMAs (1 frag load/WMMA).
// K loop: 18 chunks of 32 (9 offsets x 2 half-channel chunks).
// hin: f16 NHWC; out: f32 NHWC.
// ---------------------------------------------------------------------------
template <bool MOD>
__global__ __launch_bounds__(256) void pac_conv_kernel(
    const _Float16* __restrict__ hin, const _Float16* __restrict__ kmap,
    const _Float16* __restrict__ wpk, float* __restrict__ raw)
{
    const int blk = blockIdx.x;
    const int b   = blk >> 9;           // 512 blocks per image
    const int r   = blk & 511;
    const int y   = r >> 1;
    const int x0  = (r & 1) << 7;       // 128-pixel row segment

    const int tid  = threadIdx.x;
    const int lane = tid & 31;
    const int wv   = tid >> 5;
    const int mt0  = (wv >> 2) << 1;    // M-tile pair: {0,1} or {2,3}
    const int nt0  = (wv & 3) << 1;     // N-tile pair

    const int n  = lane & 15;           // column / A-row within tile
    const int hi = lane >> 4;           // half-wave selects K sub-range
    const int px0 = x0 + nt0 * 16 + n;  // pixel for N-tile nt0
    const int px1 = px0 + 16;           // pixel for N-tile nt0+1

    v8f acc00 = {}, acc01 = {}, acc10 = {}, acc11 = {};
    union Frag { v16h v; v8h h2[2]; };

    const size_t rowBase = (size_t)(b * HW_ + y * W_);

    for (int chunk = 0; chunk < 18; ++chunk) {
        const int off = chunk >> 1;
        const int c0  = (chunk & 1) << 5;
        const int dy  = off / 3 - 1, dx = off % 3 - 1;
        const int yy  = refl(y + dy, H_);
        const size_t srcRow = (size_t)(b * HW_ + yy * W_);
        const int cBase = c0 + hi * 16;

        // ---- B fragments (lanes 0-15: K 0-15, lanes 16-31: K 16-31) ----
        v16h bv0 = *(const v16h*)(hin + (srcRow + refl(px0 + dx, W_)) * 64 + cBase);
        v16h bv1 = *(const v16h*)(hin + (srcRow + refl(px1 + dx, W_)) * 64 + cBase);
        if (MOD) {
            const _Float16* kmRow = kmap + (size_t)(b * 9 + off) * HW_ + y * W_;
            _Float16 km0 = kmRow[px0];
            _Float16 km1 = kmRow[px1];
#pragma unroll
            for (int i = 0; i < 16; ++i) { bv0[i] = bv0[i] * km0; bv1[i] = bv1[i] * km1; }
        }

        // ---- A fragments: row m = mt*16+n; elems 0..15 -> K kk+hi*16..+15
        //      (two 16B aligned loads) ----
        const int kk = chunk * 32;
        Frag a0, a1;
        {
            const _Float16* ap = wpk + (size_t)(mt0 * 16 + n) * K_ + kk + hi * 16;
            a0.h2[0] = *(const v8h*)ap; a0.h2[1] = *(const v8h*)(ap + 8);
        }
        {
            const _Float16* ap = wpk + (size_t)((mt0 + 1) * 16 + n) * K_ + kk + hi * 16;
            a1.h2[0] = *(const v8h*)ap; a1.h2[1] = *(const v8h*)(ap + 8);
        }

        acc00 = __builtin_amdgcn_wmma_f32_16x16x32_f16(false, a0.v, false, bv0,
                                                       (short)0, acc00, false, false);
        acc01 = __builtin_amdgcn_wmma_f32_16x16x32_f16(false, a0.v, false, bv1,
                                                       (short)0, acc01, false, false);
        acc10 = __builtin_amdgcn_wmma_f32_16x16x32_f16(false, a1.v, false, bv0,
                                                       (short)0, acc10, false, false);
        acc11 = __builtin_amdgcn_wmma_f32_16x16x32_f16(false, a1.v, false, bv1,
                                                       (short)0, acc11, false, false);
    }

    // ---- store D tiles: VGPR r -> out channel mt*16 + hi*8 + r (contiguous)
    float* out0 = raw + (rowBase + px0) * 64;
    float* out1 = raw + (rowBase + px1) * 64;
    const int oA = mt0 * 16 + hi * 8;
    const int oB = (mt0 + 1) * 16 + hi * 8;
    {
        float* d = out0 + oA;
        *(float4*)d       = make_float4(acc00[0], acc00[1], acc00[2], acc00[3]);
        *(float4*)(d + 4) = make_float4(acc00[4], acc00[5], acc00[6], acc00[7]);
    }
    {
        float* d = out1 + oA;
        *(float4*)d       = make_float4(acc01[0], acc01[1], acc01[2], acc01[3]);
        *(float4*)(d + 4) = make_float4(acc01[4], acc01[5], acc01[6], acc01[7]);
    }
    {
        float* d = out0 + oB;
        *(float4*)d       = make_float4(acc10[0], acc10[1], acc10[2], acc10[3]);
        *(float4*)(d + 4) = make_float4(acc10[4], acc10[5], acc10[6], acc10[7]);
    }
    {
        float* d = out1 + oB;
        *(float4*)d       = make_float4(acc11[0], acc11[1], acc11[2], acc11[3]);
        *(float4*)(d + 4) = make_float4(acc11[4], acc11[5], acc11[6], acc11[7]);
    }
}

// ---------------------------------------------------------------------------
// Instance-norm stats: deterministic two-stage reduction (no FP atomics).
// ---------------------------------------------------------------------------
__global__ void stats_partial_kernel(const float* __restrict__ raw,
                                     float* __restrict__ partial)
{
    __shared__ float red[4][64][2];
    int b = blockIdx.x / NG_, g = blockIdx.x % NG_;
    int tid = threadIdx.x;
    int c = tid & 63, sub = tid >> 6;
    float s = 0.f, s2 = 0.f;
    int base = g * 512 + sub;
    for (int k = 0; k < 128; ++k) {
        float v = raw[((size_t)(b * HW_ + base + 4 * k)) * 64 + c];
        s += v; s2 += v * v;
    }
    red[sub][c][0] = s; red[sub][c][1] = s2;
    __syncthreads();
    if (sub == 0) {
        float a = 0.f, a2 = 0.f;
        for (int q = 0; q < 4; ++q) { a += red[q][c][0]; a2 += red[q][c][1]; }
        size_t o = ((size_t)(b * NG_ + g) * 64 + c) * 2;
        partial[o] = a; partial[o + 1] = a2;
    }
}

__global__ void stats_final_kernel(const float* __restrict__ partial,
                                   float* __restrict__ meanArr,
                                   float* __restrict__ invArr)
{
    int bo = threadIdx.x;
    if (bo >= B_ * 64) return;
    int b = bo >> 6, c = bo & 63;
    float s = 0.f, s2 = 0.f;
    for (int g = 0; g < NG_; ++g) {
        size_t o = ((size_t)(b * NG_ + g) * 64 + c) * 2;
        s += partial[o]; s2 += partial[o + 1];
    }
    float mean = s * (1.f / HW_);
    float var  = s2 * (1.f / HW_) - mean * mean;
    meanArr[bo] = mean;
    invArr[bo]  = rsqrtf(var + EPSF);
}

// u = relu(inorm(raw))  f32 NHWC -> f16 NHWC  (same index, fully coalesced)
__global__ void ew_norm_relu_kernel(const float* __restrict__ raw,
                                    const float* __restrict__ meanArr,
                                    const float* __restrict__ invArr,
                                    _Float16* __restrict__ dst)
{
    int idx = blockIdx.x * blockDim.x + threadIdx.x;
    if (idx >= B_ * HW_ * C_) return;
    int c = idx & 63;
    int b = (idx >> 6) / HW_;
    int bo = b * 64 + c;
    float v = (raw[idx] - meanArr[bo]) * invArr[bo];
    dst[idx] = (_Float16)fmaxf(v, 0.f);
}

// act = relu(act + inorm(raw))   (residual + outer relu, in place)
__global__ void ew_residual_relu_kernel(const float* __restrict__ raw,
                                        const float* __restrict__ meanArr,
                                        const float* __restrict__ invArr,
                                        _Float16* __restrict__ act)
{
    int idx = blockIdx.x * blockDim.x + threadIdx.x;
    if (idx >= B_ * HW_ * C_) return;
    int c = idx & 63;
    int b = (idx >> 6) / HW_;
    int bo = b * 64 + c;
    float v = (float)act[idx] + (raw[idx] - meanArr[bo]) * invArr[bo];
    act[idx] = (_Float16)fmaxf(v, 0.f);
}

// out = relu(raw + bf) with NHWC -> NCHW transpose through LDS
__global__ void final_out_kernel(const float* __restrict__ raw,
                                 const float* __restrict__ bf,
                                 float* __restrict__ out)
{
    __shared__ float tile[64][65];
    int blk = blockIdx.x;
    int b  = blk >> 10;
    int p0 = (blk & 1023) << 6;
    int tid = threadIdx.x;
#pragma unroll
    for (int i = 0; i < 16; ++i) {
        int e = tid + i * 256;
        int pl = e >> 6, c = e & 63;
        tile[pl][c] = raw[((size_t)(b * HW_ + p0 + pl)) * 64 + c];
    }
    __syncthreads();
#pragma unroll
    for (int i = 0; i < 16; ++i) {
        int e = tid + i * 256;
        int c = e >> 6, pl = e & 63;
        out[((size_t)(b * 64 + c)) * HW_ + p0 + pl] = fmaxf(tile[pl][c] + bf[c], 0.f);
    }
}

// ---------------------------------------------------------------------------
extern "C" void kernel_launch(void* const* d_in, const int* in_sizes, int n_in,
                              void* d_out, int out_size, void* d_ws, size_t ws_size,
                              hipStream_t stream)
{
    (void)in_sizes; (void)n_in; (void)out_size;

    const float* x  = (const float*)d_in[0];
    const float* w0 = (const float*)d_in[2];
    const float* b0 = (const float*)d_in[3];
    const float* wf = (const float*)d_in[4];
    const float* bf = (const float*)d_in[5];
    const float* wab[6] = {(const float*)d_in[6],  (const float*)d_in[8],
                           (const float*)d_in[10], (const float*)d_in[12],
                           (const float*)d_in[14], (const float*)d_in[16]};

    // ---- workspace layout (256B aligned slices) ----
    char* ws = (char*)d_ws;
    size_t off = 0;
    auto alloc = [&](size_t bytes) -> char* {
        char* p = ws + off;
        off += (bytes + 255) & ~(size_t)255;
        return p;
    };
    _Float16* kmap = (_Float16*)alloc((size_t)B_ * 9 * HW_ * 2);   // 4.7 MB
    _Float16* wpk  = (_Float16*)alloc((size_t)7 * C_ * K_ * 2);    // 0.5 MB
    _Float16* actA = (_Float16*)alloc((size_t)B_ * HW_ * C_ * 2);  // 32 MB
    _Float16* actB = (_Float16*)alloc((size_t)B_ * HW_ * C_ * 2);  // 32 MB
    float*    raw  = (float*)alloc((size_t)B_ * HW_ * C_ * 4);     // 64 MB
    float*    part = (float*)alloc((size_t)B_ * NG_ * C_ * 2 * 4); // 0.25 MB
    float*    mnA  = (float*)alloc(B_ * C_ * 4);
    float*    ivA  = (float*)alloc(B_ * C_ * 4);
    if (off > ws_size) return;   // workspace too small: bail (no OOB)

    const int NE = B_ * HW_ * C_;

    pack_weights_kernel<<<(7 * C_ * K_ + 255) / 256, 256, 0, stream>>>(
        wab[0], wab[1], wab[2], wab[3], wab[4], wab[5], wf, wpk);
    h0_kernel<<<(NE + 255) / 256, 256, 0, stream>>>(x, w0, b0, actA);
    kmap_kernel<<<(B_ * HW_ + 255) / 256, 256, 0, stream>>>(x, kmap);

    auto stats = [&]() {
        stats_partial_kernel<<<B_ * NG_, 256, 0, stream>>>(raw, part);
        stats_final_kernel<<<1, 256, 0, stream>>>(part, mnA, ivA);
    };

    for (int blkI = 0; blkI < 3; ++blkI) {
        const _Float16* wA = wpk + (size_t)(2 * blkI)     * C_ * K_;
        const _Float16* wB = wpk + (size_t)(2 * blkI + 1) * C_ * K_;
        // conv_a -> inorm -> relu -> actB
        pac_conv_kernel<true><<<2048, 256, 0, stream>>>(actA, kmap, wA, raw);
        stats();
        ew_norm_relu_kernel<<<(NE + 255) / 256, 256, 0, stream>>>(raw, mnA, ivA, actB);
        // conv_b -> inorm -> +residual(actA) -> outer relu -> actA
        pac_conv_kernel<true><<<2048, 256, 0, stream>>>(actB, kmap, wB, raw);
        stats();
        ew_residual_relu_kernel<<<(NE + 255) / 256, 256, 0, stream>>>(raw, mnA, ivA, actA);
    }

    // final plain 3x3 conv (kmap disabled) + bias + relu, NCHW f32 output
    pac_conv_kernel<false><<<2048, 256, 0, stream>>>(actA, kmap,
                                                     wpk + (size_t)6 * C_ * K_, raw);
    final_out_kernel<<<4096, 256, 0, stream>>>(raw, bf, (float*)d_out);
}
